// ResidualGNN_1889785610249
// MI455X (gfx1250) — matrix operand order repaired and checked by reading, not compiled, not run on hardware
//
#include <hip/hip_runtime.h>
#include <math.h>

#define N_NODES   100000
#define N_EDGES   1600000
#define F_IN      128
#define N_CLASSES 3
#define N_GRAPHS  16
#define LN_EPS    1e-5f

typedef __attribute__((ext_vector_type(16))) _Float16 v16h;
typedef __attribute__((ext_vector_type(8)))  _Float16 v8h;
typedef __attribute__((ext_vector_type(4)))  _Float16 v4h;
typedef __attribute__((ext_vector_type(8)))  float    v8f;

union V16 {               // build v16h from two 16-byte pieces / scalars
    v16h     v;
    v8h      h2[2];
    _Float16 e[16];
};

// ---------------------------------------------------------------- utilities
__device__ __forceinline__ float wave_sum32(float v) {
#pragma unroll
    for (int m = 16; m >= 1; m >>= 1) v += __shfl_xor(v, m, 32);
    return v;
}

// ---------------------------------------------------------------- degree
__global__ void deg_init_kernel(float* __restrict__ deg) {
    int n = blockIdx.x * blockDim.x + threadIdx.x;
    if (n < N_NODES) deg[n] = 1.0f;   // self loop
}

__global__ void deg_edge_kernel(const int* __restrict__ dst, float* __restrict__ deg) {
    int e = blockIdx.x * blockDim.x + threadIdx.x;
    if (e < N_EDGES) atomicAdd(&deg[dst[e]], 1.0f);
}

__global__ void deg_rsqrt_kernel(float* __restrict__ deg) {
    int n = blockIdx.x * blockDim.x + threadIdx.x;
    if (n < N_NODES) deg[n] = rsqrtf(deg[n]);   // deg -> deg^-1/2 in place
}

// ---------------------------------------------------------------- weight pre-swizzle into WMMA B-fragment order
// Layout: Wswz[(((ks*8 + nt)*32 + lane)*16 + i] = f16(W0[K][N]),
//   K = ks*32 + (i>>3)*16 + (lane>>4)*8 + (i&7) , N = nt*16 + (lane&15)
// so each lane's 16 halves for a (k-step, N-tile) are one contiguous 32-byte run.
__global__ void w0_swizzle_kernel(const float* __restrict__ W, _Float16* __restrict__ Wswz) {
    int idx  = blockIdx.x * blockDim.x + threadIdx.x;   // 0..16383
    int i    = idx & 15;
    int lane = (idx >> 4) & 31;
    int ntks = idx >> 9;                                // 0..31
    int nt   = ntks & 7;
    int ks   = ntks >> 3;
    int K    = ks * 32 + (i >> 3) * 16 + (lane >> 4) * 8 + (i & 7);
    int N    = nt * 16 + (lane & 15);
    Wswz[idx] = (_Float16)W[K * F_IN + N];
}

// W1 (128x3) zero-padded to 16 columns: Wswz[(ks*32 + lane)*16 + i]
__global__ void w1_swizzle_kernel(const float* __restrict__ W, _Float16* __restrict__ Wswz) {
    int idx  = blockIdx.x * blockDim.x + threadIdx.x;   // 0..2047
    int i    = idx & 15;
    int lane = (idx >> 4) & 31;
    int ks   = idx >> 9;                                // 0..3
    int K    = ks * 32 + (i >> 3) * 16 + (lane >> 4) * 8 + (i & 7);
    int l    = lane & 15;
    Wswz[idx] = (l < N_CLASSES) ? (_Float16)W[K * N_CLASSES + l] : (_Float16)0.0f;
}

// ---------------------------------------------------------------- GEMM0: h0 = x @ W0  (100000x128 @ 128x128)
// block = 256 threads = 8 waves; block handles a 32-row slab (grid 3125, exact).
// wave = N-tile; each wave computes 2 M-tiles, reusing its B fragment.
// A slab staged to LDS as f16; B fragments are contiguous global_load_b128 pairs.
__global__ void gemm0_kernel(const float* __restrict__ X, const _Float16* __restrict__ Bswz,
                             float* __restrict__ H) {
    __shared__ __align__(16) _Float16 As[32 * F_IN];    // 8 KB

    const int t    = threadIdx.x;
    const int wave = t >> 5;                            // 0..7 -> N tile
    const int lane = t & 31;
    const int half = lane >> 4;                         // 0: lanes 0-15, 1: lanes 16-31
    const int l    = lane & 15;
    const int row0 = blockIdx.x * 32;
    const int col0 = wave * 16;

    // Stage A slab: 4096 f32 -> f16, 256 threads x 16 elements
    const float* Xs = X + (size_t)row0 * F_IN;
#pragma unroll
    for (int j = 0; j < 4; ++j) {
        int base = t * 16 + j * 4;                      // 0..4092 step 4
        float4 f = *(const float4*)(Xs + base);
        v4h hv;
        hv[0] = (_Float16)f.x; hv[1] = (_Float16)f.y;
        hv[2] = (_Float16)f.z; hv[3] = (_Float16)f.w;
        *(v4h*)(&As[base]) = hv;                        // ds_store_b64
    }
    __syncthreads();

    v8f acc0 = {}, acc1 = {};
#pragma unroll
    for (int ks = 0; ks < 4; ++ks) {
        const int k0 = ks * 32;
        // B fragment: one contiguous 32-byte run per lane (2x global_load_b128)
        const _Float16* bp = Bswz + (size_t)(((ks * 8 + wave) * 32 + lane) * 16);
        V16 b;
        b.h2[0] = *(const v8h*)(bp);
        b.h2[1] = *(const v8h*)(bp + 8);
        // A fragments, 16-bit A 16x32 layout:
        //   half 0 -> K in {k0..k0+7} U {k0+16..k0+23}; half 1 -> +8
        const _Float16* a0 = As + (l)      * F_IN + k0 + half * 8;
        const _Float16* a1 = As + (16 + l) * F_IN + k0 + half * 8;
        V16 fa0, fa1;
        fa0.h2[0] = *(const v8h*)(a0);
        fa0.h2[1] = *(const v8h*)(a0 + 16);
        fa1.h2[0] = *(const v8h*)(a1);
        fa1.h2[1] = *(const v8h*)(a1 + 16);
        acc0 = __builtin_amdgcn_wmma_f32_16x16x32_f16(false, fa0.v, false, b.v,
                                                      (short)0, acc0, false, false);
        acc1 = __builtin_amdgcn_wmma_f32_16x16x32_f16(false, fa1.v, false, b.v,
                                                      (short)0, acc1, false, false);
    }

    // C/D layout: VGPR r -> M = r + half*8, N = l
    float* o0 = H + (size_t)(row0 + half * 8)      * F_IN + col0 + l;
    float* o1 = H + (size_t)(row0 + 16 + half * 8) * F_IN + col0 + l;
#pragma unroll
    for (int r = 0; r < 8; ++r) {
        o0[(size_t)r * F_IN] = acc0[r];
        o1[(size_t)r * F_IN] = acc1[r];
    }
}

// ---------------------------------------------------------------- layer-0 aggregation
__global__ void agg_init0_kernel(const float* __restrict__ h0, const float* __restrict__ dinv,
                                 const float* __restrict__ b0, float* __restrict__ out) {
    int idx  = blockIdx.x * blockDim.x + threadIdx.x;   // node*32 + group
    int node = idx >> 5;
    int grp  = (idx & 31) * 4;
    float w  = dinv[node]; w *= w;
    float4 v  = *(const float4*)(h0 + (size_t)node * F_IN + grp);
    float4 bb = *(const float4*)(b0 + grp);
    float4 o;
    o.x = v.x * w + bb.x; o.y = v.y * w + bb.y;
    o.z = v.z * w + bb.z; o.w = v.w * w + bb.w;
    *(float4*)(out + (size_t)node * F_IN + grp) = o;
}

// one wave per edge, lane = 4-feature group; L2-resident gather + float atomics scatter
__global__ void edge_agg0_kernel(const int* __restrict__ src, const int* __restrict__ dst,
                                 const float* __restrict__ h, const float* __restrict__ dinv,
                                 float* __restrict__ out) {
    int idx = blockIdx.x * blockDim.x + threadIdx.x;
    int e   = idx >> 5;
    int grp = (idx & 31) * 4;
    int s = src[e], d = dst[e];
    float w = dinv[s] * dinv[d];
    float4 v = *(const float4*)(h + (size_t)s * F_IN + grp);
    float* o = out + (size_t)d * F_IN + grp;
    atomicAdd(o + 0, v.x * w);
    atomicAdd(o + 1, v.y * w);
    atomicAdd(o + 2, v.z * w);
    atomicAdd(o + 3, v.w * w);
}

// ---------------------------------------------------------------- LayerNorm + ReLU + TopK gate, 128 feats (wave per node)
__global__ void ln_gate0_kernel(float* __restrict__ h, const float* __restrict__ g,
                                const float* __restrict__ be, const float* __restrict__ p) {
    int wid = (blockIdx.x * blockDim.x + threadIdx.x) >> 5;    // node
    int grp = (threadIdx.x & 31) * 4;
    float* row = h + (size_t)wid * F_IN + grp;
    float4 v = *(const float4*)row;

    float s  = v.x + v.y + v.z + v.w;
    float sq = v.x * v.x + v.y * v.y + v.z * v.z + v.w * v.w;
    s  = wave_sum32(s);
    sq = wave_sum32(sq);
    float mean = s * (1.0f / F_IN);
    float var  = sq * (1.0f / F_IN) - mean * mean;
    float inv  = rsqrtf(var + LN_EPS);

    float4 gg = *(const float4*)(g + grp);
    float4 bb = *(const float4*)(be + grp);
    float4 pp = *(const float4*)(p + grp);
    float4 y;
    y.x = fmaxf((v.x - mean) * inv * gg.x + bb.x, 0.0f);
    y.y = fmaxf((v.y - mean) * inv * gg.y + bb.y, 0.0f);
    y.z = fmaxf((v.z - mean) * inv * gg.z + bb.z, 0.0f);
    y.w = fmaxf((v.w - mean) * inv * gg.w + bb.w, 0.0f);

    float dot = y.x * pp.x + y.y * pp.y + y.z * pp.z + y.w * pp.w;
    float pn  = pp.x * pp.x + pp.y * pp.y + pp.z * pp.z + pp.w * pp.w;
    dot = wave_sum32(dot);
    pn  = wave_sum32(pn);
    float gate = tanhf(dot * rsqrtf(pn));
    y.x *= gate; y.y *= gate; y.z *= gate; y.w *= gate;
    *(float4*)row = y;
}

// ---------------------------------------------------------------- GEMM1: h1 = h @ W1 (N=3, zero-padded B from swizzled f16)
// block = 64 threads = 2 waves, grid = 3125 -> 6250 tiles, exact.
__global__ void gemm1_kernel(const float* __restrict__ Hin, const _Float16* __restrict__ Bswz,
                             float* __restrict__ Hout) {
    const int wave = threadIdx.x >> 5;
    const int lane = threadIdx.x & 31;
    const int half = lane >> 4;
    const int l    = lane & 15;
    const int row0 = (blockIdx.x * 2 + wave) * 16;

    v8f acc = {};
#pragma unroll
    for (int ks = 0; ks < 4; ++ks) {
        const int k0 = ks * 32;
        const float* arow = Hin + (size_t)(row0 + l) * F_IN + k0 + half * 8;
        float4 f0 = *(const float4*)(arow);
        float4 f1 = *(const float4*)(arow + 4);
        float4 f2 = *(const float4*)(arow + 16);
        float4 f3 = *(const float4*)(arow + 20);
        V16 a;
        a.e[0]  = (_Float16)f0.x; a.e[1]  = (_Float16)f0.y;
        a.e[2]  = (_Float16)f0.z; a.e[3]  = (_Float16)f0.w;
        a.e[4]  = (_Float16)f1.x; a.e[5]  = (_Float16)f1.y;
        a.e[6]  = (_Float16)f1.z; a.e[7]  = (_Float16)f1.w;
        a.e[8]  = (_Float16)f2.x; a.e[9]  = (_Float16)f2.y;
        a.e[10] = (_Float16)f2.z; a.e[11] = (_Float16)f2.w;
        a.e[12] = (_Float16)f3.x; a.e[13] = (_Float16)f3.y;
        a.e[14] = (_Float16)f3.z; a.e[15] = (_Float16)f3.w;

        const _Float16* bp = Bswz + (size_t)((ks * 32 + lane) * 16);
        V16 b;
        b.h2[0] = *(const v8h*)(bp);
        b.h2[1] = *(const v8h*)(bp + 8);

        acc = __builtin_amdgcn_wmma_f32_16x16x32_f16(false, a.v, false, b.v,
                                                     (short)0, acc, false, false);
    }
    if (l < N_CLASSES) {
#pragma unroll
        for (int r = 0; r < 8; ++r)
            Hout[(size_t)(row0 + r + half * 8) * N_CLASSES + l] = acc[r];
    }
}

// ---------------------------------------------------------------- layer-1 aggregation (3 feats)
__global__ void agg_init1_kernel(const float* __restrict__ h1, const float* __restrict__ dinv,
                                 const float* __restrict__ b1, float* __restrict__ out) {
    int n = blockIdx.x * blockDim.x + threadIdx.x;
    if (n >= N_NODES) return;
    float w = dinv[n]; w *= w;
#pragma unroll
    for (int f = 0; f < N_CLASSES; ++f)
        out[(size_t)n * N_CLASSES + f] = h1[(size_t)n * N_CLASSES + f] * w + b1[f];
}

__global__ void edge_agg1_kernel(const int* __restrict__ src, const int* __restrict__ dst,
                                 const float* __restrict__ h, const float* __restrict__ dinv,
                                 float* __restrict__ out) {
    int e = blockIdx.x * blockDim.x + threadIdx.x;
    if (e >= N_EDGES) return;
    int s = src[e], d = dst[e];
    float w = dinv[s] * dinv[d];
#pragma unroll
    for (int f = 0; f < N_CLASSES; ++f)
        atomicAdd(&out[(size_t)d * N_CLASSES + f], h[(size_t)s * N_CLASSES + f] * w);
}

// ---------------------------------------------------------------- pool scratch init
__global__ void pool_zero_kernel(float* __restrict__ gsum, float* __restrict__ gcnt) {
    int i = threadIdx.x;
    if (i < N_GRAPHS * N_CLASSES) gsum[i] = 0.0f;
    if (i < N_GRAPHS)             gcnt[i] = 0.0f;
}

// ---------------------------------------------------------------- LN + ReLU + gate (3 feats) + mean-pool accumulate
__global__ void ln_gate1_pool_kernel(const float* __restrict__ h, const float* __restrict__ g,
                                     const float* __restrict__ be, const float* __restrict__ p,
                                     const int* __restrict__ batch,
                                     float* __restrict__ gsum, float* __restrict__ gcnt) {
    int n = blockIdx.x * blockDim.x + threadIdx.x;
    if (n >= N_NODES) return;
    float a = h[(size_t)n * 3 + 0], b = h[(size_t)n * 3 + 1], c = h[(size_t)n * 3 + 2];
    float mean = (a + b + c) * (1.0f / 3.0f);
    float va = a - mean, vb = b - mean, vc = c - mean;
    float var = (va * va + vb * vb + vc * vc) * (1.0f / 3.0f);
    float inv = rsqrtf(var + LN_EPS);
    float y0 = fmaxf(va * inv * g[0] + be[0], 0.0f);
    float y1 = fmaxf(vb * inv * g[1] + be[1], 0.0f);
    float y2 = fmaxf(vc * inv * g[2] + be[2], 0.0f);
    float pn = p[0] * p[0] + p[1] * p[1] + p[2] * p[2];
    float gate = tanhf((y0 * p[0] + y1 * p[1] + y2 * p[2]) * rsqrtf(pn));
    y0 *= gate; y1 *= gate; y2 *= gate;
    int gi = batch[n];
    atomicAdd(&gsum[gi * 3 + 0], y0);
    atomicAdd(&gsum[gi * 3 + 1], y1);
    atomicAdd(&gsum[gi * 3 + 2], y2);
    atomicAdd(&gcnt[gi], 1.0f);
}

// ---------------------------------------------------------------- MLP head (16 graphs, one thread each)
__global__ void head_kernel(const float* __restrict__ gsum, const float* __restrict__ gcnt,
                            const float* __restrict__ fw1, const float* __restrict__ fb1,
                            const float* __restrict__ fw2, const float* __restrict__ fb2,
                            const float* __restrict__ fw3, const float* __restrict__ fb3,
                            float* __restrict__ out) {
    int gi = threadIdx.x;
    if (gi >= N_GRAPHS) return;
    const int H = 2 * N_CLASSES;
    float cnt = fmaxf(gcnt[gi], 1.0f);
    float gv[N_CLASSES];
#pragma unroll
    for (int i = 0; i < N_CLASSES; ++i) gv[i] = gsum[gi * N_CLASSES + i] / cnt;
    float t1[6];
#pragma unroll
    for (int o = 0; o < H; ++o) {
        float acc = fb1[o];
#pragma unroll
        for (int i = 0; i < N_CLASSES; ++i) acc += gv[i] * fw1[i * H + o];
        t1[o] = fmaxf(acc, 0.0f);
    }
    float t2[6];
#pragma unroll
    for (int o = 0; o < H; ++o) {
        float acc = fb2[o];
#pragma unroll
        for (int i = 0; i < H; ++i) acc += t1[i] * fw2[i * H + o];
        t2[o] = fmaxf(acc, 0.0f);
    }
#pragma unroll
    for (int o = 0; o < N_CLASSES; ++o) {
        float acc = fb3[o];
#pragma unroll
        for (int i = 0; i < H; ++i) acc += t2[i] * fw3[i * N_CLASSES + o];
        out[gi * N_CLASSES + o] = tanhf(acc);
    }
}

// ---------------------------------------------------------------- launch
extern "C" void kernel_launch(void* const* d_in, const int* in_sizes, int n_in,
                              void* d_out, int out_size, void* d_ws, size_t ws_size,
                              hipStream_t stream) {
    const float* x    = (const float*)d_in[0];
    const int*   ei   = (const int*)  d_in[1];
    const int*   batch= (const int*)  d_in[2];
    const float* W0   = (const float*)d_in[3];
    const float* b0   = (const float*)d_in[4];
    const float* g0   = (const float*)d_in[5];
    const float* be0  = (const float*)d_in[6];
    const float* p0   = (const float*)d_in[7];
    const float* W1   = (const float*)d_in[8];
    const float* b1   = (const float*)d_in[9];
    const float* g1   = (const float*)d_in[10];
    const float* be1  = (const float*)d_in[11];
    const float* p1   = (const float*)d_in[12];
    const float* fw1  = (const float*)d_in[13];
    const float* fb1  = (const float*)d_in[14];
    const float* fw2  = (const float*)d_in[15];
    const float* fb2  = (const float*)d_in[16];
    const float* fw3  = (const float*)d_in[17];
    const float* fb3  = (const float*)d_in[18];

    const int* src = ei;
    const int* dst = ei + N_EDGES;

    // workspace carve (floats, 256B-aligned chunks)
    float* ws   = (float*)d_ws;
    float* deg  = ws;                                   // 100000  (becomes deg^-1/2)
    float* h0   = ws + 100352;                          // 12.8M
    float* agg0 = h0 + 12800000;                        // 12.8M
    float* h1   = agg0 + 12800000;                      // 300K
    float* agg1 = h1 + 300032;                          // 300K
    float* gsum = agg1 + 300032;                        // 48
    float* gcnt = gsum + 64;                            // 16
    _Float16* w0swz = (_Float16*)(gsum + 128);          // 16384 halves (32 KB)
    _Float16* w1swz = (_Float16*)(gsum + 128 + 8192);   // 2048 halves  (4 KB)
    float* out  = (float*)d_out;                        // 16 x 3

    const int nb_nodes = (N_NODES + 255) / 256;         // 391
    const int nb_edges = (N_EDGES + 255) / 256;         // 6250

    // degrees -> deg^-1/2 ; weight swizzles
    deg_init_kernel  <<<nb_nodes, 256, 0, stream>>>(deg);
    deg_edge_kernel  <<<nb_edges, 256, 0, stream>>>(dst, deg);
    deg_rsqrt_kernel <<<nb_nodes, 256, 0, stream>>>(deg);
    w0_swizzle_kernel<<<64, 256, 0, stream>>>(W0, w0swz);
    w1_swizzle_kernel<<<8,  256, 0, stream>>>(W1, w1swz);

    // layer 0
    gemm0_kernel    <<<N_NODES / 32, 256, 0, stream>>>(x, w0swz, h0);            // 3125, exact
    agg_init0_kernel<<<N_NODES * 32 / 256, 256, 0, stream>>>(h0, deg, b0, agg0); // 12500, exact
    edge_agg0_kernel<<<N_EDGES * 32 / 256, 256, 0, stream>>>(src, dst, h0, deg, agg0); // 200000, exact
    ln_gate0_kernel <<<N_NODES * 32 / 256, 256, 0, stream>>>(agg0, g0, be0, p0); // 12500, exact

    // layer 1
    gemm1_kernel    <<<N_NODES / 32, 64, 0, stream>>>(agg0, w1swz, h1);          // 3125, exact
    agg_init1_kernel<<<nb_nodes, 256, 0, stream>>>(h1, deg, b1, agg1);
    edge_agg1_kernel<<<nb_edges, 256, 0, stream>>>(src, dst, h1, deg, agg1);

    // pooling + head
    pool_zero_kernel    <<<1, 64, 0, stream>>>(gsum, gcnt);
    ln_gate1_pool_kernel<<<nb_nodes, 256, 0, stream>>>(agg1, g1, be1, p1, batch, gsum, gcnt);
    head_kernel         <<<1, 32, 0, stream>>>(gsum, gcnt, fw1, fb1, fw2, fb2, fw3, fb3, out);
}
